// RegionProposalNetwork_70566312673640
// MI455X (gfx1250) — matrix operand order repaired and checked by reading, not compiled, not run on hardware
//
#include <hip/hip_runtime.h>

// ---------------- types ----------------
typedef __attribute__((ext_vector_type(16))) _Float16 v16h;
typedef __attribute__((ext_vector_type(8)))  _Float16 v8h;
typedef __attribute__((ext_vector_type(8)))  float    v8f;

#define HH 64
#define WW 64
#define CC 512
#define PH 66
#define NIMG 4
#define NANCH 36864       // 64*64*9
#define NSORT 65536
#define NPRE 6000
#define NPOST 300

// ---------------- prep: pad+convert input to f16 NHWC ----------------
__global__ void prep_x(const float* __restrict__ x, _Float16* __restrict__ xh) {
    size_t i = (size_t)blockIdx.x * blockDim.x + threadIdx.x;
    const size_t tot = (size_t)NIMG * PH * PH * CC;
    if (i >= tot) return;
    int ci = (int)(i % CC); size_t t = i / CC;
    int xx = (int)(t % PH); t /= PH;
    int yy = (int)(t % PH); int img = (int)(t / PH);
    float v = 0.f;
    if (yy >= 1 && yy <= HH && xx >= 1 && xx <= WW)
        v = x[(((size_t)img * CC + ci) * HH + (yy - 1)) * WW + (xx - 1)];
    xh[i] = (_Float16)v;
}

// conv1_w OIHW (512,512,3,3) -> wB[tap][co][ci] f16
__global__ void prep_w(const float* __restrict__ w, _Float16* __restrict__ wB) {
    size_t i = (size_t)blockIdx.x * blockDim.x + threadIdx.x;
    const size_t tot = (size_t)9 * CC * CC;
    if (i >= tot) return;
    int ci = (int)(i % CC); size_t t = i / CC;
    int co = (int)(t % CC); int tap = (int)(t / CC);
    wB[i] = (_Float16)w[((size_t)co * CC + ci) * 9 + tap];
}

// fuse loc_w (36,512) + score_w (18,512) -> wSL[64][512] f16, biases -> bSL[64]
__global__ void prep_wsl(const float* __restrict__ loc_w, const float* __restrict__ score_w,
                         const float* __restrict__ loc_b, const float* __restrict__ score_b,
                         _Float16* __restrict__ wSL, float* __restrict__ bSL) {
    int i = blockIdx.x * blockDim.x + threadIdx.x;
    if (i >= 64 * CC) return;
    int ci = i % CC, co = i / CC;
    float v = 0.f;
    if (co < 36) v = loc_w[(size_t)co * CC + ci];
    else if (co < 54) v = score_w[(size_t)(co - 36) * CC + ci];
    wSL[i] = (_Float16)v;
    if (ci == 0) bSL[co] = (co < 36) ? loc_b[co] : ((co < 54) ? score_b[co - 36] : 0.f);
}

// ---------------- anchors ----------------
__global__ void anchors_k(float* __restrict__ anchor) {
    int i = blockIdx.x * blockDim.x + threadIdx.x;
    if (i >= NANCH) return;
    int a = i % 9; int p = i / 9; int x = p % WW; int y = p / WW;
    const float R[3] = {0.5f, 1.f, 2.f};
    const float S[3] = {8.f, 16.f, 32.f};
    int r = a / 3, s = a % 3;
    float h = 16.f * S[s] * sqrtf(R[r]);
    float w = 16.f * S[s] * sqrtf(1.f / R[r]);
    float sy = y * 16.f, sx = x * 16.f;
    anchor[(size_t)i * 4 + 0] = sy + 8.f - 0.5f * h;
    anchor[(size_t)i * 4 + 1] = sx + 8.f - 0.5f * w;
    anchor[(size_t)i * 4 + 2] = sy + 8.f + 0.5f * h;
    anchor[(size_t)i * 4 + 3] = sx + 8.f + 0.5f * w;
}

// ---------------- 3x3 conv + bias + ReLU via WMMA ----------------
// grid: (64 rows, 2 co-blocks of 256, 4 imgs), block 256 = 8 waves
// wave tile: 32 spatial x 64 co  => 8 WMMAs per 12 b128 fragment loads
__global__ __launch_bounds__(256) void conv3x3_wmma(
    const _Float16* __restrict__ xh,   // [4][66][66][512]
    const _Float16* __restrict__ wB,   // [9][512][512]  (tap, co, ci)
    const float*    __restrict__ bias, // [512]
    _Float16*       __restrict__ feat) // [4][64][64][512]
{
    const int lane = threadIdx.x & 31;
    const int wv   = threadIdx.x >> 5;
    const int y    = blockIdx.x;
    const int img  = blockIdx.z;
    const int s0   = (wv & 1) * 32;                      // spatial x base
    const int co0  = blockIdx.y * 256 + (wv >> 1) * 64;  // co base
    const int m    = lane & 15;
    const int hi   = lane >> 4;

    v8f acc[2][4] = {};

    for (int tap = 0; tap < 9; ++tap) {
        const int ky = tap / 3, kx = tap % 3;
        const _Float16* xrow = xh + ((size_t)img * PH + (y + ky)) * PH * CC;
        const _Float16* wtap = wB + (size_t)tap * CC * CC;
        if (tap < 8) {
            // warm L0/L2 for the next tap's input row (global_prefetch_b8)
            const int nky = (tap + 1) / 3, nkx = (tap + 1) % 3;
            const _Float16* nrow = xh + ((size_t)img * PH + (y + nky)) * PH * CC
                                      + (size_t)(s0 + 2 * lane + nkx) * CC;
            __builtin_prefetch(nrow, 0, 1);
        }
        for (int cb = 0; cb < CC; cb += 32) {
            v16h a[2], b[4];
#pragma unroll
            for (int i = 0; i < 2; ++i) {
                const _Float16* pa = xrow + (size_t)(s0 + 16 * i + m + kx) * CC + cb;
                v8h lo = *(const v8h*)(pa + hi * 8);
                v8h hg = *(const v8h*)(pa + 16 + hi * 8);
                a[i] = __builtin_shufflevector(lo, hg, 0,1,2,3,4,5,6,7,8,9,10,11,12,13,14,15);
            }
#pragma unroll
            for (int j = 0; j < 4; ++j) {
                const _Float16* pb = wtap + (size_t)(co0 + 16 * j + m) * CC + cb + hi * 8;
                v8h lo = *(const v8h*)(pb);
                v8h hg = *(const v8h*)(pb + 16);
                b[j] = __builtin_shufflevector(lo, hg, 0,1,2,3,4,5,6,7,8,9,10,11,12,13,14,15);
            }
#pragma unroll
            for (int i = 0; i < 2; ++i)
#pragma unroll
                for (int j = 0; j < 4; ++j)
                    acc[i][j] = __builtin_amdgcn_wmma_f32_16x16x32_f16(
                        false, a[i], false, b[j], (short)0, acc[i][j], false, false);
        }
    }
#pragma unroll
    for (int j = 0; j < 4; ++j) {
        float bj = bias[co0 + 16 * j + m];
#pragma unroll
        for (int i = 0; i < 2; ++i)
#pragma unroll
            for (int r = 0; r < 8; ++r) {
                float v = acc[i][j][r] + bj;
                v = v > 0.f ? v : 0.f;
                int xo = s0 + 16 * i + r + 8 * hi;
                feat[(((size_t)img * HH + y) * WW + xo) * CC + co0 + 16 * j + m] = (_Float16)v;
            }
    }
}

// ---------------- 1x1 convs (score 18 + loc 36, padded to 64) via WMMA ----------------
// grid: 128 blocks of 256 threads; wave covers 16 spatial x 64 co
__global__ __launch_bounds__(256) void conv1x1_wmma(
    const _Float16* __restrict__ feat,   // [16384][512]
    const _Float16* __restrict__ wSL,    // [64][512]
    const float*    __restrict__ bSL,    // [64]
    float* __restrict__ out_locs,        // [16384*36]
    float* __restrict__ out_scores)      // [16384*18]
{
    const int lane = threadIdx.x & 31;
    const int wv   = threadIdx.x >> 5;
    const int m    = lane & 15;
    const int hi   = lane >> 4;
    const long s0  = (long)blockIdx.x * 128 + wv * 16;

    v8f acc[4] = {};
    for (int cb = 0; cb < CC; cb += 32) {
        const _Float16* pa = feat + (size_t)(s0 + m) * CC + cb;
        v8h lo = *(const v8h*)(pa + hi * 8);
        v8h hg = *(const v8h*)(pa + 16 + hi * 8);
        v16h a = __builtin_shufflevector(lo, hg, 0,1,2,3,4,5,6,7,8,9,10,11,12,13,14,15);
#pragma unroll
        for (int j = 0; j < 4; ++j) {
            const _Float16* pb = wSL + (size_t)(16 * j + m) * CC + cb + hi * 8;
            v8h blo = *(const v8h*)(pb);
            v8h bhg = *(const v8h*)(pb + 16);
            v16h b = __builtin_shufflevector(blo, bhg, 0,1,2,3,4,5,6,7,8,9,10,11,12,13,14,15);
            acc[j] = __builtin_amdgcn_wmma_f32_16x16x32_f16(
                false, a, false, b, (short)0, acc[j], false, false);
        }
    }
#pragma unroll
    for (int j = 0; j < 4; ++j) {
        int co = 16 * j + m;
        float bj = bSL[co];
#pragma unroll
        for (int r = 0; r < 8; ++r) {
            long s = s0 + r + 8 * hi;
            float v = acc[j][r] + bj;
            if (co < 36)      out_locs[s * 36 + co] = v;
            else if (co < 54) out_scores[s * 18 + (co - 36)] = v;
        }
    }
}

// ---------------- softmax fg = sigmoid(s1 - s0) ----------------
__global__ void fg_k(const float* __restrict__ scores, float* __restrict__ fg) {
    int i = blockIdx.x * blockDim.x + threadIdx.x;
    if (i >= NIMG * NANCH) return;
    float s0 = scores[2 * (size_t)i], s1 = scores[2 * (size_t)i + 1];
    fg[i] = 1.f / (1.f + expf(s0 - s1));
}

// ---------------- decode + clip + min-size + pack sort keys ----------------
__global__ void decode_k(const float* __restrict__ locs, const float* __restrict__ fg,
                         const float* __restrict__ anchor,
                         const int* __restrict__ ph, const int* __restrict__ pw,
                         float* __restrict__ boxes, unsigned long long* __restrict__ keys) {
    int img = blockIdx.y;
    int i = blockIdx.x * blockDim.x + threadIdx.x;
    if (i >= NSORT) return;
    unsigned long long* K = keys + (size_t)img * NSORT;
    if (i >= NANCH) { K[i] = 0ull; return; }
    float imgh = (float)ph[0], imgw = (float)pw[0];
    const float* A = anchor + (size_t)i * 4;
    const float* L = locs + (size_t)img * (NANCH * 4) + (size_t)i * 4;
    float h = A[2] - A[0], w = A[3] - A[1];
    float cy = A[0] + 0.5f * h, cx = A[1] + 0.5f * w;
    float ncy = L[0] * h + cy, ncx = L[1] * w + cx;
    float nh = expf(L[2]) * h, nw = expf(L[3]) * w;
    float y0 = fminf(fmaxf(ncy - 0.5f * nh, 0.f), imgh);
    float x0 = fminf(fmaxf(ncx - 0.5f * nw, 0.f), imgw);
    float y1 = fminf(fmaxf(ncy + 0.5f * nh, 0.f), imgh);
    float x1 = fminf(fmaxf(ncx + 0.5f * nw, 0.f), imgw);
    float* B = boxes + ((size_t)img * NANCH + i) * 4;
    B[0] = y0; B[1] = x0; B[2] = y1; B[3] = x1;
    bool valid = (y1 - y0 >= 16.f) && (x1 - x0 >= 16.f);
    float sc = valid ? fg[(size_t)img * NANCH + i] : -1e30f;
    unsigned u = __float_as_uint(sc);
    u = (u & 0x80000000u) ? ~u : (u | 0x80000000u);   // order-preserving flip
    K[i] = ((unsigned long long)u << 32) | (unsigned)i;
}

// ---------------- bitonic sort pass (descending) ----------------
__global__ void bitonic_pass(unsigned long long* __restrict__ keys, int j, int k) {
    int img = blockIdx.y;
    unsigned long long* K = keys + (size_t)img * NSORT;
    int i = blockIdx.x * blockDim.x + threadIdx.x;
    int ixj = i ^ j;
    if (ixj <= i) return;
    bool up = ((i & k) == 0);     // descending overall
    unsigned long long a = K[i], b = K[ixj];
    bool sw = up ? (a < b) : (a > b);
    if (sw) { K[i] = b; K[ixj] = a; }
}

// ---------------- greedy NMS on sorted pool ----------------
__global__ __launch_bounds__(256) void nms_k(const unsigned long long* __restrict__ keys,
                                             const float* __restrict__ boxes,
                                             float* __restrict__ rois) {
    int img = blockIdx.x;
    const unsigned long long* K = keys + (size_t)img * NSORT;
    const float* B = boxes + (size_t)img * NANCH * 4;
    float* R = rois + (size_t)img * NPOST * 4;
    __shared__ float kb[NPOST][4];
    __shared__ int nk;
    __shared__ int sup;
    for (int t = threadIdx.x; t < NPOST * 4; t += blockDim.x) R[t] = 0.f;
    if (threadIdx.x == 0) nk = 0;
    __syncthreads();
    unsigned ut = ~__float_as_uint(-5e29f);   // flip(NEG_INF/2), negative -> ~u
    for (int c = 0; c < NPRE; ++c) {
        unsigned long long key = K[c];
        unsigned uk = (unsigned)(key >> 32);
        if (uk <= ut) break;                  // pool exhausted of valid scores
        int idx = (int)(key & 0xFFFFFFFFull);
        float y0 = B[(size_t)idx * 4 + 0], x0 = B[(size_t)idx * 4 + 1];
        float y1 = B[(size_t)idx * 4 + 2], x1 = B[(size_t)idx * 4 + 3];
        if (threadIdx.x == 0) sup = 0;
        __syncthreads();
        float area = (y1 - y0) * (x1 - x0);
        int n = nk;
        for (int t = threadIdx.x; t < n; t += blockDim.x) {
            float ty0 = kb[t][0], tx0 = kb[t][1], ty1 = kb[t][2], tx1 = kb[t][3];
            float ih = fmaxf(fminf(y1, ty1) - fmaxf(y0, ty0), 0.f);
            float iw = fmaxf(fminf(x1, tx1) - fmaxf(x0, tx0), 0.f);
            float inter = ih * iw;
            float ta = (ty1 - ty0) * (tx1 - tx0);
            float iou = inter / fmaxf(area + ta - inter, 1e-6f);
            if (iou > 0.7f) sup = 1;
        }
        __syncthreads();
        if (sup == 0 && threadIdx.x == 0) {
            kb[nk][0] = y0; kb[nk][1] = x0; kb[nk][2] = y1; kb[nk][3] = x1;
            R[(size_t)nk * 4 + 0] = y0; R[(size_t)nk * 4 + 1] = x0;
            R[(size_t)nk * 4 + 2] = y1; R[(size_t)nk * 4 + 3] = x1;
            nk++;
        }
        __syncthreads();
        if (nk >= NPOST) break;
    }
}

// ---------------- launch ----------------
extern "C" void kernel_launch(void* const* d_in, const int* in_sizes, int n_in,
                              void* d_out, int out_size, void* d_ws, size_t ws_size,
                              hipStream_t stream) {
    const float* x       = (const float*)d_in[0];
    const float* conv1_w = (const float*)d_in[1];
    const float* conv1_b = (const float*)d_in[2];
    const float* score_w = (const float*)d_in[3];
    const float* score_b = (const float*)d_in[4];
    const float* loc_w   = (const float*)d_in[5];
    const float* loc_b   = (const float*)d_in[6];
    const int*   img_h   = (const int*)d_in[7];
    const int*   img_w   = (const int*)d_in[8];

    float* out        = (float*)d_out;
    float* out_locs   = out;                       // 4*36864*4
    float* out_scores = out + 589824;              // 4*36864*2
    float* out_rois   = out + 884736;              // 4*300*4
    float* out_anchor = out + 889536;              // 36864*4

    char* ws = (char*)d_ws;
    size_t off = 0;
    auto alloc = [&](size_t bytes) -> void* {
        void* p = ws + off;
        off += (bytes + 255) & ~(size_t)255;
        return p;
    };
    _Float16* xh   = (_Float16*)alloc((size_t)NIMG * PH * PH * CC * 2);
    _Float16* wB   = (_Float16*)alloc((size_t)9 * CC * CC * 2);
    _Float16* wSL  = (_Float16*)alloc((size_t)64 * CC * 2);
    float*    bSL  = (float*)alloc(64 * 4);
    _Float16* feat = (_Float16*)alloc((size_t)NIMG * HH * WW * CC * 2);
    float*    fg   = (float*)alloc((size_t)NIMG * NANCH * 4);
    float*    bx   = (float*)alloc((size_t)NIMG * NANCH * 16);
    unsigned long long* keys = (unsigned long long*)alloc((size_t)NIMG * NSORT * 8);

    {
        size_t tot = (size_t)NIMG * PH * PH * CC;
        prep_x<<<dim3((unsigned)((tot + 255) / 256)), 256, 0, stream>>>(x, xh);
    }
    prep_w<<<(9 * CC * CC + 255) / 256, 256, 0, stream>>>(conv1_w, wB);
    prep_wsl<<<(64 * CC + 255) / 256, 256, 0, stream>>>(loc_w, score_w, loc_b, score_b, wSL, bSL);
    anchors_k<<<(NANCH + 255) / 256, 256, 0, stream>>>(out_anchor);

    conv3x3_wmma<<<dim3(HH, 2, NIMG), 256, 0, stream>>>(xh, wB, conv1_b, feat);
    conv1x1_wmma<<<128, 256, 0, stream>>>(feat, wSL, bSL, out_locs, out_scores);

    fg_k<<<(NIMG * NANCH + 255) / 256, 256, 0, stream>>>(out_scores, fg);
    decode_k<<<dim3(NSORT / 256, NIMG), 256, 0, stream>>>(out_locs, fg, out_anchor,
                                                          img_h, img_w, bx, keys);
    for (int k = 2; k <= NSORT; k <<= 1)
        for (int j = k >> 1; j > 0; j >>= 1)
            bitonic_pass<<<dim3(NSORT / 256, NIMG), 256, 0, stream>>>(keys, j, k);
    nms_k<<<NIMG, 256, 0, stream>>>(keys, bx, out_rois);

    (void)in_sizes; (void)n_in; (void)out_size; (void)ws_size;
}
